// LSTM_82824149336295
// MI455X (gfx1250) — matrix operand order repaired and checked by reading, not compiled, not run on hardware
//
#include <hip/hip_runtime.h>

// ---------------------------------------------------------------------------
// LSTM, S=512 B=128 I=256 H=512, gates (f,i,o,c) stacked along 4H.
// Persistent-kernel design for MI455X (gfx1250, wave32, WMMA):
//   * one fused GEMM per step:  g = [x_t | h] * [W;U]^T   (K = 768)
//   * v_wmma_f32_16x16x32_bf16, f32 accumulate
//   * 32 blocks x 256 threads = 256 waves; each wave owns one 16x16 cell tile
//     (c-state in registers), computes 4 gate tiles per step
//   * A operand (shared by all 8 waves of a block) staged once per step into
//     LDS via GLOBAL_LOAD_ASYNC_TO_LDS_B128 (ASYNCcnt), read via ds_load
//   * bf16 operands pre-packed into the exact ISA per-lane A/B layouts
//   * double-buffered bf16 h in workspace + atomic grid barrier per step
// ---------------------------------------------------------------------------

typedef __attribute__((ext_vector_type(16))) __bf16 v16bf;
typedef __attribute__((ext_vector_type(8)))  float  v8f;
typedef __attribute__((address_space(3))) char lds_char;

#define Sd   512
#define Bd   128
#define Id   256
#define Hd   512
#define KTOT 768              // I + H
#define NKCH 24               // KTOT / 32
#define NXCH 8                // I / 32
#define NHCH 16               // H / 32

// workspace layout (bytes)
#define BAR_OFF   0                       // 2 x u32 barrier state
#define HBF_OFF   256                     // 2 buffers of h in bf16 A-layout
#define HBF_ELEMS 65536                   // 8 rt * 16 kk * 32 lane * 16 e
#define BBF_OFF   (HBF_OFF + 2 * HBF_ELEMS * 2)          // packed [W;U] bf16
#define BBF_GRPS  (128 * NKCH)            // 128 col-tiles * 24 k-chunks
#define XBF_OFF   (BBF_OFF + (size_t)BBF_GRPS * 512 * 2) // packed x bf16
#define XBF_GRPS  (Sd * 8 * NXCH)         // 32768 groups of 32 lanes * 16 e

// issue one 16-byte async global->LDS copy (per lane), tracked by ASYNCcnt
__device__ __forceinline__ void async_g2l_b128(unsigned lds_off,
                                               const void* gaddr) {
  unsigned long long ga = (unsigned long long)(uintptr_t)gaddr;
  asm volatile("global_load_async_to_lds_b128 %0, %1, off"
               :: "v"(lds_off), "v"(ga) : "memory");
}
__device__ __forceinline__ void wait_asynccnt0() {
  asm volatile("s_wait_asynccnt 0x0" ::: "memory");
}

// --------------------------- init: barrier + h=0 ---------------------------
__global__ void lstm_init(unsigned* __restrict__ bar,
                          unsigned* __restrict__ hbf_u32) {
  int i = blockIdx.x * blockDim.x + threadIdx.x;
  if (i < 2) bar[i] = 0u;
  if (i < HBF_ELEMS) hbf_u32[i] = 0u;   // zero BOTH bf16 h buffers (as dwords)
}

// ------------------- pack [W;U] -> bf16, WMMA B-layout ---------------------
// B 32x16 bf16 layout: lane n(0..15) = col n, elems 0..7 -> K kk*32+half*8+e,
//                      elems 8..15 -> K kk*32+16+half*8+(e-8); half=lane>>4.
__global__ void lstm_pack_w(const float* __restrict__ W,
                            const float* __restrict__ U,
                            __bf16* __restrict__ Bw) {
  int gid  = blockIdx.x * blockDim.x + threadIdx.x;   // 98304 threads
  int lane = gid & 31;
  int grp  = gid >> 5;                                // < 3072
  int ct   = grp / NKCH;                              // col tile (0..127)
  int kk   = grp % NKCH;                              // k chunk  (0..23)
  int col  = ct * 16 + (lane & 15);                   // gate col (0..2047)
  int kbase = kk * 32 + (lane >> 4) * 8;
  v16bf o;
#pragma unroll
  for (int e = 0; e < 16; ++e) {
    int k = kbase + e + ((e < 8) ? 0 : 8);
    float f = (k < Id) ? W[(size_t)col * Id + k]
                       : U[(size_t)col * Hd + (k - Id)];
    o[e] = (__bf16)f;
  }
  *(v16bf*)(Bw + (size_t)grp * 512 + lane * 16) = o;
}

// ---------------------- pack x -> bf16, WMMA A-layout ----------------------
// A 16x32 bf16 layout: lane L(0..15) = row L, elems 0..7 -> K kk*32+half*8+e,
//                      elems 8..15 -> K kk*32+16+half*8+(e-8); half=lane>>4.
__global__ void lstm_pack_x(const float* __restrict__ x,
                            __bf16* __restrict__ xbf) {
  int gid  = blockIdx.x * blockDim.x + threadIdx.x;   // 1048576 threads
  int lane = gid & 31;
  int grp  = gid >> 5;                                // < 32768
  int t    = grp >> 6;
  int rem  = grp & 63;
  int rt   = rem >> 3;                                // batch row-tile 0..7
  int kk   = rem & 7;                                 // x k-chunk 0..7
  int row  = rt * 16 + (lane & 15);
  int half = lane >> 4;
  const float* src = x + ((size_t)t * Bd + row) * Id;
  v16bf o;
#pragma unroll
  for (int e = 0; e < 16; ++e) {
    int k = (e < 8) ? (kk * 32 + half * 8 + e)
                    : (kk * 32 + 16 + half * 8 + (e - 8));
    o[e] = (__bf16)src[k];
  }
  *(v16bf*)(xbf + (size_t)grp * 512 + lane * 16) = o;
}

// ------------------------- persistent LSTM kernel --------------------------
__launch_bounds__(256, 1)
__global__ void lstm_persistent(const float* __restrict__ bW,
                                const float* __restrict__ bU,
                                float* __restrict__ out,
                                unsigned* __restrict__ bar,
                                __bf16* __restrict__ hbf,
                                const __bf16* __restrict__ Bw,
                                const __bf16* __restrict__ xbf) {
  // A-operand staging buffer: 24 chunks x 1KB (shared by the 8 waves)
  __shared__ __align__(16) char ldsA[NKCH * 1024];

  const int lane  = threadIdx.x & 31;
  const int wv    = threadIdx.x >> 5;
  const int rt    = blockIdx.x & 7;       // batch row-tile (16 rows)
  const int cg    = blockIdx.x >> 3;      // col group
  const int ht    = cg * 8 + wv;          // h col-tile (0..31)
  const int b0    = rt * 16;
  const int h0    = ht * 16;
  const int n     = lane & 15;            // D-layout: N column
  const int mbase = (lane >> 4) * 8;      // D-layout: first M row

  // bias = bW + bU per gate, per column
  const float biasF = bW[0 * Hd + h0 + n] + bU[0 * Hd + h0 + n];
  const float biasI = bW[1 * Hd + h0 + n] + bU[1 * Hd + h0 + n];
  const float biasO = bW[2 * Hd + h0 + n] + bU[2 * Hd + h0 + n];
  const float biasC = bW[3 * Hd + h0 + n] + bU[3 * Hd + h0 + n];

  // constants for scattering h into the bf16 A-layout state buffer
  const int gk   = h0 + n;                // this lane's global h column
  const int kk2  = gk >> 5;               // target k-chunk
  const int kc   = gk & 31;
  const int half = (kc >> 3) & 1;
  const int ee   = (kc & 7) + 8 * (kc >> 4);

  // B-operand tile bases (gate order f,i,o,c => col tile = g*32 + ht)
  const size_t ctF = (size_t)(0 * 32 + ht) * NKCH;
  const size_t ctI = (size_t)(1 * 32 + ht) * NKCH;
  const size_t ctO = (size_t)(2 * 32 + ht) * NKCH;
  const size_t ctC = (size_t)(3 * 32 + ht) * NKCH;

  // this wave stages chunks [wv*3, wv*3+3); per-lane 32B -> 2x b128 copies
  const unsigned ldsBase =
      (unsigned)(uintptr_t)(lds_char*)(ldsA) + (unsigned)(wv * 3) * 1024u
      + (unsigned)lane * 32u;

  float cst[8];
#pragma unroll
  for (int v = 0; v < 8; ++v) cst[v] = 0.0f;

  for (int t = 0; t < Sd; ++t) {
    const __bf16* hcur  = hbf + (size_t)(t & 1) * HBF_ELEMS;
    __bf16*       hnext = hbf + (size_t)((t + 1) & 1) * HBF_ELEMS;

    // ---- stage the 24 A chunks (x_t then h) into LDS, async ----
#pragma unroll
    for (int c = 0; c < 3; ++c) {
      int kk = wv * 3 + c;
      const char* g = (kk < NXCH)
          ? (const char*)(xbf + ((((size_t)t * 8 + rt) * NXCH + kk) * 32) * 16)
          : (const char*)(hcur + (((size_t)(rt * 16 + (kk - NXCH)) * 32) * 16));
      g += lane * 32;
      unsigned l = ldsBase + (unsigned)c * 1024u;
      async_g2l_b128(l, g);
      async_g2l_b128(l + 16u, g + 16);
    }
    wait_asynccnt0();
    __syncthreads();

    // ---- 24 K-chunks: A from LDS (ds_load), B from global (L2-resident) ----
    v8f aF = {}, aI = {}, aO = {}, aC = {};
#pragma unroll 4
    for (int kk = 0; kk < NKCH; ++kk) {
      v16bf a = *(const v16bf*)(ldsA + kk * 1024 + lane * 32);
      v16bf bF = *(const v16bf*)(Bw + ((ctF + kk) * 32 + lane) * 16);
      v16bf bI = *(const v16bf*)(Bw + ((ctI + kk) * 32 + lane) * 16);
      v16bf bO = *(const v16bf*)(Bw + ((ctO + kk) * 32 + lane) * 16);
      v16bf bC = *(const v16bf*)(Bw + ((ctC + kk) * 32 + lane) * 16);
      aF = __builtin_amdgcn_wmma_f32_16x16x32_bf16(false, a, false, bF, (short)0, aF, false, false);
      aI = __builtin_amdgcn_wmma_f32_16x16x32_bf16(false, a, false, bI, (short)0, aI, false, false);
      aO = __builtin_amdgcn_wmma_f32_16x16x32_bf16(false, a, false, bO, (short)0, aO, false, false);
      aC = __builtin_amdgcn_wmma_f32_16x16x32_bf16(false, a, false, bC, (short)0, aC, false, false);
    }

    // ---- gate nonlinearity + cell update (c in registers) ----
    float* hs = out + (size_t)t * Bd * Hd;
#pragma unroll
    for (int v = 0; v < 8; ++v) {
      float gF = aF[v] + biasF;
      float gI = aI[v] + biasI;
      float gO = aO[v] + biasO;
      float gC = aC[v] + biasC;
      float f  = 1.0f / (1.0f + __expf(-gF));
      float i  = 1.0f / (1.0f + __expf(-gI));
      float o  = 1.0f / (1.0f + __expf(-gO));
      float cn = f * cst[v] + i * tanhf(gC);
      cst[v]   = cn;
      float h  = o * tanhf(cn);

      int row = mbase + v;
      hs[(size_t)(b0 + row) * Hd + h0 + n] = h;                 // h_seq (f32)
      // scatter into next-step bf16 A-layout state
      hnext[((rt * 16 + kk2) * 32 + (row + 16 * half)) * 16 + ee] = (__bf16)h;
      if (t == Sd - 1) {
        size_t base = (size_t)Sd * Bd * Hd;
        out[base + (size_t)(b0 + row) * Hd + h0 + n] = h;       // h_final
        out[base + (size_t)Bd * Hd + (size_t)(b0 + row) * Hd + h0 + n] = cn; // c_final
      }
    }

    // ---- grid-wide barrier (32 blocks): release h, acquire next step ----
    __syncthreads();
    if (threadIdx.x == 0) {
      __threadfence();
      unsigned g = __hip_atomic_load(&bar[1], __ATOMIC_RELAXED,
                                     __HIP_MEMORY_SCOPE_AGENT);
      unsigned prev = __hip_atomic_fetch_add(&bar[0], 1u, __ATOMIC_ACQ_REL,
                                             __HIP_MEMORY_SCOPE_AGENT);
      if (prev == 31u) {
        __hip_atomic_store(&bar[0], 0u, __ATOMIC_RELAXED,
                           __HIP_MEMORY_SCOPE_AGENT);
        __hip_atomic_fetch_add(&bar[1], 1u, __ATOMIC_RELEASE,
                               __HIP_MEMORY_SCOPE_AGENT);
      } else {
        while (__hip_atomic_load(&bar[1], __ATOMIC_ACQUIRE,
                                 __HIP_MEMORY_SCOPE_AGENT) == g)
          __builtin_amdgcn_s_sleep(2);
      }
    }
    __syncthreads();
  }
}

// ---------------------------------------------------------------------------
extern "C" void kernel_launch(void* const* d_in, const int* in_sizes, int n_in,
                              void* d_out, int out_size, void* d_ws, size_t ws_size,
                              hipStream_t stream) {
  const float* x  = (const float*)d_in[0];   // [S,B,I]
  const float* W  = (const float*)d_in[1];   // [4H,I]
  const float* bW = (const float*)d_in[2];   // [4H]
  const float* U  = (const float*)d_in[3];   // [4H,H]
  const float* bU = (const float*)d_in[4];   // [4H]
  float* out = (float*)d_out;                // h_seq | h_fin | c_fin

  char* ws = (char*)d_ws;
  unsigned* bar = (unsigned*)(ws + BAR_OFF);
  __bf16*   hbf = (__bf16*)(ws + HBF_OFF);
  __bf16*   Bw  = (__bf16*)(ws + BBF_OFF);
  __bf16*   xbf = (__bf16*)(ws + XBF_OFF);

  lstm_init<<<256, 256, 0, stream>>>(bar, (unsigned*)hbf);
  lstm_pack_w<<<(BBF_GRPS * 32) / 256, 256, 0, stream>>>(W, U, Bw);
  lstm_pack_x<<<(XBF_GRPS * 32) / 256, 256, 0, stream>>>(x, xbf);
  lstm_persistent<<<32, 256, 0, stream>>>(bW, bU, out, bar, hbf, Bw, xbf);
}
